// Community2Emb_61546881352240
// MI455X (gfx1250) — matrix operand order repaired and checked by reading, not compiled, not run on hardware
//
#include <hip/hip_runtime.h>
#include <hip/hip_bf16.h>
#include <math.h>

#define N_PTS 20000
#define DIM   128
#define KC    32
#define NT    (N_PTS / 16)          // 1250 n-tiles of 16 points
#define WPB   10                    // waves (n-tiles) per block
#define BLOCKS_PER_K (NT / WPB)     // 125

typedef __attribute__((ext_vector_type(16))) __bf16 v16bf;
typedef __attribute__((ext_vector_type(8)))  __bf16 v8bf;
typedef __attribute__((ext_vector_type(8)))  float  v8f;
typedef __attribute__((ext_vector_type(4)))  unsigned int u32x4;
typedef __attribute__((ext_vector_type(8)))  int i32x8;
typedef __attribute__((ext_vector_type(4)))  int i32x4;

// ---------------------------------------------------------------------------
// Kernel 1: per-component Cholesky (in LDS) + logdet.  32 blocks x 128 thr.
// ---------------------------------------------------------------------------
__global__ __launch_bounds__(DIM) void chol_logdet_kernel(
    const float* __restrict__ cov, float* __restrict__ Lg,
    float* __restrict__ logdet)
{
  __shared__ float Ls[DIM][DIM];                 // 64 KB (WGP has 320 KB LDS)
  const int k = blockIdx.x, t = threadIdx.x;
  const float* A = cov + (size_t)k * DIM * DIM;
  for (int j = 0; j < DIM; ++j) Ls[t][j] = A[t * DIM + j];
  __syncthreads();

  for (int j = 0; j < DIM; ++j) {                // left-looking, in place
    if (t == j) {
      float s = Ls[j][j];
      for (int p = 0; p < j; ++p) s -= Ls[j][p] * Ls[j][p];
      Ls[j][j] = sqrtf(s);
    }
    __syncthreads();
    if (t > j) {
      float s = Ls[t][j];
      for (int p = 0; p < j; ++p) s -= Ls[t][p] * Ls[j][p];
      Ls[t][j] = s / Ls[j][j];
    }
    __syncthreads();
  }

  float* Lk = Lg + (size_t)k * DIM * DIM;
  for (int j = 0; j < DIM; ++j) Lk[t * DIM + j] = (j <= t) ? Ls[t][j] : 0.0f;
  if (t == 0) {
    float s = 0.0f;
    for (int j = 0; j < DIM; ++j) s += logf(Ls[j][j]);
    logdet[k] = 2.0f * s;
  }
}

// ---------------------------------------------------------------------------
// Kernel 2: W = L^{-1} (lower triangular), stored as hi/lo bf16 split.
// Thread c computes column c by forward substitution. 32 blocks x 128 thr.
// ---------------------------------------------------------------------------
__global__ __launch_bounds__(DIM) void trinv_split_kernel(
    const float* __restrict__ Lg, __bf16* __restrict__ Whi,
    __bf16* __restrict__ Wlo)
{
  const int k = blockIdx.x, c = threadIdx.x;
  const float* Lk = Lg + (size_t)k * DIM * DIM;
  __bf16* Wh = Whi + (size_t)k * DIM * DIM;
  __bf16* Wl = Wlo + (size_t)k * DIM * DIM;

  for (int i = 0; i < c; ++i) {                  // strictly-upper part = 0
    Wh[i * DIM + c] = (__bf16)0.0f;
    Wl[i * DIM + c] = (__bf16)0.0f;
  }
  {
    float w = 1.0f / Lk[c * DIM + c];
    __bf16 h = (__bf16)w;
    Wh[c * DIM + c] = h;
    Wl[c * DIM + c] = (__bf16)(w - (float)h);
  }
  for (int i = c + 1; i < DIM; ++i) {
    float s = 0.0f;
    for (int p = c; p < i; ++p)
      s += Lk[i * DIM + p] *
           ((float)Wh[p * DIM + c] + (float)Wl[p * DIM + c]);
    float w = -s / Lk[i * DIM + i];
    __bf16 h = (__bf16)w;
    Wh[i * DIM + c] = h;
    Wl[i * DIM + c] = (__bf16)(w - (float)h);
  }
}

// ---------------------------------------------------------------------------
// TDM: DMA a contiguous block of `nelem` 2-byte elements from global memory
// into LDS, expressed as a 2D tensor/tile of shape [1, nelem].
// D# bitfields per CDNA5 ISA §8.3/§8.4 (group0: count/lds/global/type,
// group1: data_size + dims/strides). Groups 2/3 zero (<=2D tensor).
// ---------------------------------------------------------------------------
__device__ __forceinline__ void tdm_load_b16_block(
    unsigned int lds_off, const void* gsrc, unsigned int nelem)
{
  unsigned long long ga = (unsigned long long)(uintptr_t)gsrc;
  u32x4 g0;
  g0[0] = 1u;                                        // count=1 (valid D#)
  g0[1] = lds_off;                                   // lds_addr  [63:32]
  g0[2] = (unsigned int)ga;                          // global_addr[31:0]
  g0[3] = (unsigned int)((ga >> 32) & 0x1FFFFFFu)    // global_addr[56:32]
          | (2u << 30);                              // type=2 ("image")
  i32x8 g1;
  g1[0] = (int)(1u << 16);                           // data_size=1 -> 2 bytes
  g1[1] = (int)((nelem & 0xFFFFu) << 16);            // tensor_dim0[15:0]
  g1[2] = (int)((nelem >> 16) | (1u << 16));         // tensor_dim0[31:16], tensor_dim1=1
  g1[3] = (int)((nelem & 0xFFFFu) << 16);            // tensor_dim1 hi=0, tile_dim0=nelem
  g1[4] = 1;                                         // tile_dim1=1, tile_dim2=0
  g1[5] = (int)nelem;                                // tensor_dim0_stride[31:0]
  g1[6] = 0;                                         // stride hi / dim1_stride lo
  g1[7] = 0;
  i32x4 z4 = {0, 0, 0, 0};
#if __clang_major__ >= 23
  i32x8 z8 = {0, 0, 0, 0, 0, 0, 0, 0};
  __builtin_amdgcn_tensor_load_to_lds(g0, g1, z4, z4, z8, 0);
#else
  __builtin_amdgcn_tensor_load_to_lds(g0, g1, z4, z4, 0);
#endif
}

// ---------------------------------------------------------------------------
// Kernel 3: block = one component k x 10 n-tiles (10 waves / 320 threads).
// W_k (hi/lo bf16, 64KB) is staged into LDS once per block via the Tensor
// Data Mover; A-fragments then come from ds_load_b128. Per wave:
// z = W_k (x - mu_k) via split-bf16 WMMA, maha = sum_d z^2, fold pi/logdet.
// ---------------------------------------------------------------------------
__global__ __launch_bounds__(32 * WPB) void gmm_loss_kernel(
    const float* __restrict__ x, const float* __restrict__ centroid,
    const float* __restrict__ pi, const float* __restrict__ lambda2,
    const __bf16* __restrict__ Whi, const __bf16* __restrict__ Wlo,
    const float* __restrict__ logdet, float* __restrict__ out)
{
  __shared__ __align__(16) __bf16 shWh[DIM * DIM];   // 32 KB
  __shared__ __align__(16) __bf16 shWl[DIM * DIM];   // 32 KB

  const int lane = threadIdx.x & 31;
  const int wave = threadIdx.x >> 5;
  const int k    = blockIdx.x / BLOCKS_PER_K;        // uniform per block
  const int tb   = blockIdx.x - k * BLOCKS_PER_K;
  const int nt   = tb * WPB + wave;                  // this wave's n-tile
  const int half = lane >> 4;
  const int lm   = lane & 15;
  const int n    = nt * 16 + lm;                     // B column (point id)

  // ---- stage W_k into LDS with the TDM (one wave issues; all waves sync)
  if (wave == 0) {
    tdm_load_b16_block((unsigned int)(uintptr_t)&shWh[0],
                       Whi + (size_t)k * DIM * DIM, DIM * DIM);
    tdm_load_b16_block((unsigned int)(uintptr_t)&shWl[0],
                       Wlo + (size_t)k * DIM * DIM, DIM * DIM);
    __builtin_amdgcn_s_wait_tensorcnt(0);
  }
  __syncthreads();

  const float* xrow = x + (size_t)n * DIM;
  const float* mu   = centroid + k * DIM;

  v8f acc[8];
  #pragma unroll
  for (int dt = 0; dt < 8; ++dt) { v8f z = {}; acc[dt] = z; }

  #pragma unroll
  for (int jc = 0; jc < 4; ++jc) {                   // K-chunks of 32 along D
    const int j0 = jc * 32;

    // B fragment (32x16 bf16): lane = column n; elems = K rows j0+half*16+e.
    const int jb = j0 + half * 16;
    v8f x0 = *(const v8f*)(xrow + jb);
    v8f x1 = *(const v8f*)(xrow + jb + 8);
    v16bf Bhi, Blo;
    #pragma unroll
    for (int e = 0; e < 8; ++e) {
      float d0 = x0[e] - mu[jb + e];
      __bf16 h0 = (__bf16)d0;
      Bhi[e] = h0;  Blo[e] = (__bf16)(d0 - (float)h0);
      float d1 = x1[e] - mu[jb + 8 + e];
      __bf16 h1 = (__bf16)d1;
      Bhi[8 + e] = h1;  Blo[8 + e] = (__bf16)(d1 - (float)h1);
    }

    #pragma unroll
    for (int dt = 0; dt < 8; ++dt) {                 // 8 row-tiles of W
      // A fragment (16x32 bf16) from LDS: row d = dt*16+lm;
      // elems 0..15 -> K=j0+half*16+e.
      const int d = dt * 16 + lm;
      const __bf16* wr = shWh + d * DIM + j0;
      const __bf16* wl = shWl + d * DIM + j0;
      v8bf a0 = *(const v8bf*)(wr + half * 16);
      v8bf a1 = *(const v8bf*)(wr + half * 16 + 8);
      v8bf l0 = *(const v8bf*)(wl + half * 16);
      v8bf l1 = *(const v8bf*)(wl + half * 16 + 8);
      v16bf Ahi, Alo;
      #pragma unroll
      for (int e = 0; e < 8; ++e) {
        Ahi[e] = a0[e];  Ahi[8 + e] = a1[e];
        Alo[e] = l0[e];  Alo[8 + e] = l1[e];
      }
      // split product: (Ah+Al)(Bh+Bl) ~= Ah*Bh + Ah*Bl + Al*Bh  (f32 accum)
      acc[dt] = __builtin_amdgcn_wmma_f32_16x16x32_bf16(
          false, Ahi, false, Bhi, (short)0, acc[dt], false, false);
      acc[dt] = __builtin_amdgcn_wmma_f32_16x16x32_bf16(
          false, Ahi, false, Blo, (short)0, acc[dt], false, false);
      acc[dt] = __builtin_amdgcn_wmma_f32_16x16x32_bf16(
          false, Alo, false, Bhi, (short)0, acc[dt], false, false);
    }
  }

  // maha[n] = sum_d z^2 : per-lane partial over its 64 rows, then fold halves.
  float psum = 0.0f;
  #pragma unroll
  for (int dt = 0; dt < 8; ++dt)
    #pragma unroll
    for (int r = 0; r < 8; ++r) psum += acc[dt][r] * acc[dt][r];
  float maha = psum + __shfl_xor(psum, 16, 32);

  const float LOG2PI = 1.8378770664093453f;
  float logpdf  = -0.5f * ((float)DIM * LOG2PI + logdet[k] + maha);
  float contrib = (-lambda2[0] / (float)KC) * pi[n * KC + k] * logpdf;
  if (half == 0) atomicAdd(&out[n], contrib);        // after all WMMAs
}

// ---------------------------------------------------------------------------
extern "C" void kernel_launch(void* const* d_in, const int* in_sizes, int n_in,
                              void* d_out, int out_size, void* d_ws,
                              size_t ws_size, hipStream_t stream) {
  const float* x        = (const float*)d_in[0];   // [N,D]
  const float* centroid = (const float*)d_in[1];   // [K,D]
  const float* cov      = (const float*)d_in[2];   // [K,D,D]
  const float* pi       = (const float*)d_in[3];   // [N,K]
  const float* lambda2  = (const float*)d_in[4];   // scalar
  float* out = (float*)d_out;                      // [N]

  // workspace layout: L (2MB f32) | Whi (1MB bf16) | Wlo (1MB bf16) | logdet
  char* ws = (char*)d_ws;
  const size_t MAT = (size_t)KC * DIM * DIM;       // 524288 elements
  float*  Lg     = (float*)ws;
  __bf16* Whi    = (__bf16*)(ws + MAT * 4);
  __bf16* Wlo    = (__bf16*)(ws + MAT * 4 + MAT * 2);
  float*  logdet = (float*)(ws + MAT * 4 + MAT * 4);

  hipMemsetAsync(d_out, 0, (size_t)out_size * sizeof(float), stream);
  chol_logdet_kernel<<<KC, DIM, 0, stream>>>(cov, Lg, logdet);
  trinv_split_kernel<<<KC, DIM, 0, stream>>>(Lg, Whi, Wlo);
  gmm_loss_kernel<<<KC * BLOCKS_PER_K, 32 * WPB, 0, stream>>>(
      x, centroid, pi, lambda2, Whi, Wlo, logdet, out);
}
